// AttentionMap_69973607186633
// MI455X (gfx1250) — matrix operand order repaired and checked by reading, not compiled, not run on hardware
//
#include <hip/hip_runtime.h>

typedef float    v2f  __attribute__((ext_vector_type(2)));
typedef float    v4f  __attribute__((ext_vector_type(4)));
typedef float    v8f  __attribute__((ext_vector_type(8)));
typedef _Float16 v4h  __attribute__((ext_vector_type(4)));
typedef _Float16 v8h  __attribute__((ext_vector_type(8)));
typedef _Float16 v16h __attribute__((ext_vector_type(16)));

#define B_  4
#define H_  16
#define S_  2048
#define D_  64
#define RD_ 32

// Base-2 softmax: fold log2(e) into the logit scales so every exponential is a bare v_exp_f32.
#define LOG2E_     1.4426950408889634f
#define SCALE2_    (0.125f * LOG2E_)                 // 64^-0.5 * log2(e)
#define RELSCALE2_ (0.17677669529663687f * LOG2E_)   // 32^-0.5 * log2(e)

__device__ __forceinline__ float clamp5(float x) {
    return __builtin_amdgcn_fmed3f(x, -5.0f, 5.0f);  // single v_med3_num_f32
}

// Raw v_exp_f32. All call sites pass x <= 0: exact in [-126,0], underflows to ~0
// below (exactly what a softmax tail needs). Avoids the 6-op OCML expansion.
__device__ __forceinline__ float exp2_raw(float x) {
    return __builtin_amdgcn_exp2f(x);
}

// ---------------------------------------------------------------------------
// Preprocess: clamp to [-5,5], optional scale, convert f32 -> f16 into d_ws.
// ~150 MB of traffic total (~6 us) — removes all med3/cvt from the hot loops.
// ---------------------------------------------------------------------------
__global__ __launch_bounds__(256)
void clamp_cvt_kernel(const float* __restrict__ in, _Float16* __restrict__ out,
                      float scale, int n) {
    int i = (blockIdx.x * blockDim.x + threadIdx.x) * 4;
    const int stride = gridDim.x * blockDim.x * 4;
    for (; i < n; i += stride) {
        v4f v = *(const v4f*)(in + i);
        v4h h;
        h.x = (_Float16)(clamp5(v.x) * scale);
        h.y = (_Float16)(clamp5(v.y) * scale);
        h.z = (_Float16)(clamp5(v.z) * scale);
        h.w = (_Float16)(clamp5(v.w) * scale);
        *(v4h*)(out + i) = h;
    }
}

// ---------------------------------------------------------------------------
// f16 WMMA tile: 16x16 logit tile via 3x V_WMMA_F32_16X16X32_F16
// (2 chunks of K=32 for D=64, 1 chunk for RD=32). f32 accumulation.
// ---------------------------------------------------------------------------
__device__ __forceinline__ v8f score_tile_f16(const _Float16* __restrict__ Kb,
                                              const _Float16* __restrict__ PKb,
                                              const v16h* qa, v16h pqa) {
    v8f acc = {0.f,0.f,0.f,0.f,0.f,0.f,0.f,0.f};
#pragma unroll
    for (int cc = 0; cc < 2; ++cc) {
        v8h b0 = *(const v8h*)(Kb + cc * 32);
        v8h b1 = *(const v8h*)(Kb + cc * 32 + 16);
        v16h b = __builtin_shufflevector(b0, b1, 0,1,2,3,4,5,6,7,8,9,10,11,12,13,14,15);
        acc = __builtin_amdgcn_wmma_f32_16x16x32_f16(
            /*neg_a=*/false, qa[cc], /*neg_b=*/false, b,
            /*c_mod=*/(short)0, acc, /*reuse_a=*/false, /*reuse_b=*/false);
    }
    {
        v8h b0 = *(const v8h*)(PKb);
        v8h b1 = *(const v8h*)(PKb + 16);
        v16h b = __builtin_shufflevector(b0, b1, 0,1,2,3,4,5,6,7,8,9,10,11,12,13,14,15);
        acc = __builtin_amdgcn_wmma_f32_16x16x32_f16(
            false, pqa, false, b, (short)0, acc, false, false);
    }
    return acc;
}

__device__ __forceinline__ v16h load_a16(const _Float16* base) {
    v8h lo = *(const v8h*)(base);
    v8h hi = *(const v8h*)(base + 16);
    return __builtin_shufflevector(lo, hi, 0,1,2,3,4,5,6,7,8,9,10,11,12,13,14,15);
}

__global__ __launch_bounds__(256)
void attnmap_f16_kernel(const _Float16* __restrict__ wk,
                        const _Float16* __restrict__ wq,
                        const _Float16* __restrict__ wpk,
                        const _Float16* __restrict__ wpq,
                        float* __restrict__ out) {
    const int tid  = threadIdx.x;
    const int wave = tid >> 5;
    const int lane = tid & 31;
    const int half = lane >> 4;
    const int l16  = lane & 15;

    const int qblocks = S_ / 128;
    const int bh = blockIdx.x / qblocks;
    const int q0 = (blockIdx.x % qblocks) * 128 + wave * 16;

    // ---- A operands (pre-clamped, pre-scaled f16): Q 16x64, PQ 16x32 ----
    const size_t qrow = (size_t)bh * S_ + (size_t)(q0 + l16);
    const _Float16* Qb  = wq  + qrow * D_  + 8 * half;   // A: K = 8*half+{0..7}, +16 more
    const _Float16* PQb = wpq + qrow * RD_ + 8 * half;

    v16h qa[2];
    qa[0] = load_a16(Qb);
    qa[1] = load_a16(Qb + 32);
    v16h pqa = load_a16(PQb);

    // ---- B operand bases: lane holds Kmat[row=l16][8*half+{0..7}, 16+8*half+{0..7}] per 32-chunk ----
    const _Float16* Kbase  = wk  + ((size_t)bh * S_ + l16) * D_  + 8 * half;
    const _Float16* PKbase = wpk + ((size_t)bh * S_ + l16) * RD_ + 8 * half;

    // ---- Pass 1: per-lane streaming (max, sum); one raw v_exp per element ----
    float run_max[8], run_sum[8];
#pragma unroll
    for (int r = 0; r < 8; ++r) { run_max[r] = -3.0e38f; run_sum[r] = 0.0f; }

    for (int kt = 0; kt < S_ / 16; ++kt) {
        const _Float16* Kb  = Kbase  + (size_t)kt * 16 * D_;
        const _Float16* PKb = PKbase + (size_t)kt * 16 * RD_;
        if (kt + 1 < S_ / 16) {
            __builtin_prefetch(Kb  + 16 * D_,  0, 3);   // locality 3 -> WGP-scope prefetch
            __builtin_prefetch(PKb + 16 * RD_, 0, 3);
        }
        v8f acc = score_tile_f16(Kb, PKb, qa, pqa);
#pragma unroll
        for (int r = 0; r < 8; ++r) {
            const float v  = acc[r];
            const float rm = run_max[r];
            // nm = max(rm,v); exp2(rm-nm)/exp2(v-nm): one is 1, other exp2(-|v-rm|)
            const float e  = exp2_raw(-fabsf(v - rm));  // neg+abs fold into v_exp src mods
            const bool  gt = v > rm;
            run_sum[r] = run_sum[r] * (gt ? e : 1.0f) + (gt ? 1.0f : e);
            run_max[r] = gt ? v : rm;                   // reuse compare: v_cndmask, no canon
        }
    }

    // ---- One butterfly merge of (max,sum) across the 16 lanes per row ----
#pragma unroll
    for (int r = 0; r < 8; ++r) {
        float m = run_max[r], s = run_sum[r];
#pragma unroll
        for (int mask = 1; mask < 16; mask <<= 1) {
            const float om = __shfl_xor(m, mask, 16);
            const float os = __shfl_xor(s, mask, 16);
            const bool  gt = om > m;
            const float nm = gt ? om : m;
            s = s * exp2_raw(m - nm) + os * exp2_raw(om - nm);
            m = nm;
        }
        run_max[r] = m;
        run_sum[r] = 1.0f / s;
    }

    // ---- Pass 2: recompute tiles (L2-resident f16 K/PK) and emit probs ----
    const size_t out_row0 = ((size_t)bh * S_ + (size_t)(q0 + 8 * half)) * S_ + l16;

    for (int kt = 0; kt < S_ / 16; ++kt) {
        const _Float16* Kb  = Kbase  + (size_t)kt * 16 * D_;
        const _Float16* PKb = PKbase + (size_t)kt * 16 * RD_;
        v8f acc = score_tile_f16(Kb, PKb, qa, pqa);
#pragma unroll
        for (int r = 0; r < 8; ++r) {
            const float p = exp2_raw(acc[r] - run_max[r]) * run_sum[r];
            __builtin_nontemporal_store(p, out + out_row0 + (size_t)r * S_ + (size_t)kt * 16);
        }
    }
}

// ---------------------------------------------------------------------------
// Fallback: pure-f32 WMMA kernel (exact), used only if d_ws is too small.
// ---------------------------------------------------------------------------
__device__ __forceinline__ v8f score_tile_f32(const float* __restrict__ Kb,
                                              const float* __restrict__ PKb,
                                              const v2f* qa, const v2f* pqa) {
    v8f acc = {0.f,0.f,0.f,0.f,0.f,0.f,0.f,0.f};
#pragma unroll
    for (int c = 0; c < 16; ++c) {
        v2f kb = *(const v2f*)(Kb + c * 4);
        kb.x = clamp5(kb.x); kb.y = clamp5(kb.y);
        acc = __builtin_amdgcn_wmma_f32_16x16x4_f32(false, qa[c], false, kb,
                                                    (short)0, acc, false, false);
    }
#pragma unroll
    for (int c = 0; c < 8; ++c) {
        v2f pb = *(const v2f*)(PKb + c * 4);
        pb.x = clamp5(pb.x); pb.y = clamp5(pb.y);
        acc = __builtin_amdgcn_wmma_f32_16x16x4_f32(false, pqa[c], false, pb,
                                                    (short)0, acc, false, false);
    }
    return acc;
}

__global__ __launch_bounds__(256)
void attnmap_f32_kernel(const float* __restrict__ keys,
                        const float* __restrict__ queries,
                        const float* __restrict__ pos_key,
                        const float* __restrict__ pos_query,
                        float* __restrict__ out) {
    const int tid  = threadIdx.x;
    const int wave = tid >> 5;
    const int lane = tid & 31;
    const int half = lane >> 4;
    const int l16  = lane & 15;

    const int qblocks = S_ / 128;
    const int bh = blockIdx.x / qblocks;
    const int q0 = (blockIdx.x % qblocks) * 128 + wave * 16;

    const size_t qrow = (size_t)bh * S_ + (size_t)(q0 + l16);
    const float* Qb  = queries   + qrow * D_  + 2 * half;
    const float* PQb = pos_query + qrow * RD_ + 2 * half;

    v2f qa[16], pqa[8];
#pragma unroll
    for (int c = 0; c < 16; ++c) {
        v2f v = *(const v2f*)(Qb + c * 4);
        qa[c].x = clamp5(v.x) * SCALE2_;
        qa[c].y = clamp5(v.y) * SCALE2_;
    }
#pragma unroll
    for (int c = 0; c < 8; ++c) {
        v2f v = *(const v2f*)(PQb + c * 4);
        pqa[c].x = clamp5(v.x) * RELSCALE2_;
        pqa[c].y = clamp5(v.y) * RELSCALE2_;
    }

    const float* Kbase  = keys    + ((size_t)bh * S_ + l16) * D_  + 2 * half;
    const float* PKbase = pos_key + ((size_t)bh * S_ + l16) * RD_ + 2 * half;

    float run_max[8], run_sum[8];
#pragma unroll
    for (int r = 0; r < 8; ++r) { run_max[r] = -3.0e38f; run_sum[r] = 0.0f; }

    for (int kt = 0; kt < S_ / 16; ++kt) {
        const float* Kb  = Kbase  + (size_t)kt * 16 * D_;
        const float* PKb = PKbase + (size_t)kt * 16 * RD_;
        v8f acc = score_tile_f32(Kb, PKb, qa, pqa);
#pragma unroll
        for (int r = 0; r < 8; ++r) {
            const float v  = acc[r];
            const float rm = run_max[r];
            const float e  = exp2_raw(-fabsf(v - rm));
            const bool  gt = v > rm;
            run_sum[r] = run_sum[r] * (gt ? e : 1.0f) + (gt ? 1.0f : e);
            run_max[r] = gt ? v : rm;
        }
    }
#pragma unroll
    for (int r = 0; r < 8; ++r) {
        float m = run_max[r], s = run_sum[r];
#pragma unroll
        for (int mask = 1; mask < 16; mask <<= 1) {
            const float om = __shfl_xor(m, mask, 16);
            const float os = __shfl_xor(s, mask, 16);
            const bool  gt = om > m;
            const float nm = gt ? om : m;
            s = s * exp2_raw(m - nm) + os * exp2_raw(om - nm);
            m = nm;
        }
        run_max[r] = m;
        run_sum[r] = 1.0f / s;
    }

    const size_t out_row0 = ((size_t)bh * S_ + (size_t)(q0 + 8 * half)) * S_ + l16;
    for (int kt = 0; kt < S_ / 16; ++kt) {
        const float* Kb  = Kbase  + (size_t)kt * 16 * D_;
        const float* PKb = PKbase + (size_t)kt * 16 * RD_;
        v8f acc = score_tile_f32(Kb, PKb, qa, pqa);
#pragma unroll
        for (int r = 0; r < 8; ++r) {
            const float p = exp2_raw(acc[r] - run_max[r]) * run_sum[r];
            __builtin_nontemporal_store(p, out + out_row0 + (size_t)r * S_ + (size_t)kt * 16);
        }
    }
}

extern "C" void kernel_launch(void* const* d_in, const int* in_sizes, int n_in,
                              void* d_out, int out_size, void* d_ws, size_t ws_size,
                              hipStream_t stream) {
    (void)in_sizes; (void)n_in; (void)out_size;
    const float* keys      = (const float*)d_in[0];
    const float* queries   = (const float*)d_in[1];
    const float* pos_key   = (const float*)d_in[2];
    const float* pos_query = (const float*)d_in[3];
    float* out = (float*)d_out;

    const size_t NK = (size_t)B_ * H_ * S_ * D_;    // 8,388,608
    const size_t NP = (size_t)B_ * H_ * S_ * RD_;   // 4,194,304
    const size_t need = (2 * NK + 2 * NP) * sizeof(_Float16);  // ~50.3 MB
    const int blocks = B_ * H_ * (S_ / 128);        // 1024 blocks x 8 waves

    if (ws_size >= need) {
        _Float16* wsk  = (_Float16*)d_ws;
        _Float16* wsq  = wsk + NK;
        _Float16* wspk = wsq + NK;
        _Float16* wspq = wspk + NP;
        clamp_cvt_kernel<<<(int)(NK / 1024), 256, 0, stream>>>(keys,      wsk,  1.0f,       (int)NK);
        clamp_cvt_kernel<<<(int)(NK / 1024), 256, 0, stream>>>(queries,   wsq,  SCALE2_,    (int)NK);
        clamp_cvt_kernel<<<(int)(NP / 1024), 256, 0, stream>>>(pos_key,   wspk, 1.0f,       (int)NP);
        clamp_cvt_kernel<<<(int)(NP / 1024), 256, 0, stream>>>(pos_query, wspq, RELSCALE2_, (int)NP);
        attnmap_f16_kernel<<<blocks, 256, 0, stream>>>(wsk, wsq, wspk, wspq, out);
    } else {
        attnmap_f32_kernel<<<blocks, 256, 0, stream>>>(keys, queries, pos_key, pos_query, out);
    }
}
